// LSTM_22265110463169
// MI455X (gfx1250) — compile-verified, matrix operand order checked by reading, not run on hardware
//
#include <hip/hip_runtime.h>

// LSTM: B=2048, T=512, F=64, H=128, gates 4H=512, order i,f,g,o.
// Fully fused: per-step x@Wx + h@Wh via bf16 WMMA with ALL weight
// B-fragments register-resident; cell state c in VGPRs; h ping-pong in LDS;
// single barrier per timestep; x(t+1) prefetched into registers.

#define BDIM 2048
#define TDIM 512
#define FDIM 64
#define HDIM 128
#define G4H  512

// padded K-strides (bf16 elements) for bank-conflict-free b128 LDS loads
#define LDX 72    // K=64  (x)
#define LDH 136   // K=128 (h)

typedef __attribute__((ext_vector_type(16))) __bf16 v16bf;
typedef __attribute__((ext_vector_type(8)))  float  v8f;

union Frag {
    v16bf  v;
    __bf16 h[16];
    uint4  u[2];
};

union BF4 {
    __bf16 b[4];
    uint2  u;
};

__device__ __forceinline__ __bf16 f2bf(float x) {
    union { float f; unsigned u; } a; a.f = x;
    unsigned r = a.u + 0x7FFFu + ((a.u >> 16) & 1u);   // round-to-nearest-even
    union { unsigned short s; __bf16 b; } o; o.s = (unsigned short)(r >> 16);
    return o.b;
}

__device__ __forceinline__ float fast_sigmoid(float x) {
    float e = __builtin_amdgcn_exp2f(-1.4426950408889634f * x);   // exp(-x)
    return __builtin_amdgcn_rcpf(1.0f + e);
}
__device__ __forceinline__ float fast_tanh(float x) {
    float e = __builtin_amdgcn_exp2f(-2.8853900817779268f * x);   // exp(-2x)
    return (1.0f - e) * __builtin_amdgcn_rcpf(1.0f + e);
}

// A-fragment, 16x32 bf16, from LDS tile [16 rows][ldk K] (row = M = lane&15).
// ISA layout: lanes 0-15: K=0..7 then 16..23; lanes 16-31: K=8..15 then 24..31.
__device__ __forceinline__ v16bf load_afrag_lds(const __bf16* base, int ldk,
                                                int kbase, int lane) {
    int m  = lane & 15;
    int kh = ((lane >> 4) & 1) * 8;
    const __bf16* p = base + m * ldk + kbase + kh;
    Frag f;
    f.u[0] = *(const uint4*)(p);
    f.u[1] = *(const uint4*)(p + 16);
    return f.v;
}

// B-fragment (32x16, lane = output column N, symmetric K pattern to A)
// gathered/transposed once from a row-major f32 weight matrix W[K][ldn];
// lives in VGPRs for the whole kernel.
__device__ __forceinline__ v16bf load_bfrag_global(const float* __restrict__ W,
                                                   int ldn, int n, int kbase,
                                                   int lane) {
    int kh = ((lane >> 4) & 1) * 8;
    Frag f;
#pragma unroll
    for (int j = 0; j < 8; ++j) {
        f.h[j]     = f2bf(W[(size_t)(kbase + kh + j)      * ldn + n]);
        f.h[8 + j] = f2bf(W[(size_t)(kbase + 16 + kh + j) * ldn + n]);
    }
    return f.v;
}

__global__ void __launch_bounds__(256, 1)
lstm_fused_wmma(const float* __restrict__ inp,   // [B,T,F]
                const float* __restrict__ Wx,    // [F,4H]
                const float* __restrict__ Wh,    // [H,4H]
                const float* __restrict__ bias,  // [4H]
                float* __restrict__ out)         // [B,T,H]
{
    __shared__ __bf16 xbuf[2][16 * LDX];   // ping-pong x_t tile (bf16)
    __shared__ __bf16 hbuf[2][16 * LDH];   // ping-pong h state (bf16)

    const int tid  = threadIdx.x;
    const int lane = tid & 31;
    const int wv   = tid >> 5;              // wave 0..7 -> h-cols [16w,16w+16)
    const int b0   = blockIdx.x * 16;       // 16 batch rows per workgroup

    // ---- h(t=0) = 0 (buffer 0 is read at t=0) ----
    for (int idx = tid; idx < 16 * LDH; idx += 256) hbuf[0][idx] = f2bf(0.0f);

    // ---- all weight B-fragments gathered once -> VGPR-resident ----
    const int ncol = lane & 15;
    v16bf bh[4][4];                         // Wh: 4 gates x K{0,32,64,96}
    v16bf bx[4][2];                         // Wx: 4 gates x K{0,32}
    float bg[4];
#pragma unroll
    for (int g = 0; g < 4; ++g) {
        int cg = g * HDIM + 16 * wv;        // gate-g column base for this wave
        int n  = cg + ncol;
        bg[g]  = bias[n];
#pragma unroll
        for (int kb = 0; kb < 4; ++kb)
            bh[g][kb] = load_bfrag_global(Wh, G4H, n, kb * 32, lane);
#pragma unroll
        for (int kb = 0; kb < 2; ++kb)
            bx[g][kb] = load_bfrag_global(Wx, G4H, n, kb * 32, lane);
    }

    float cst[8];
#pragma unroll
    for (int r = 0; r < 8; ++r) cst[r] = 0.0f;

    const int mhi  = (lane >> 4) & 1;       // C-layout: lanes 16-31 -> rows 8..15
    const int hcol = 16 * wv + ncol;
    float* outp = out + (size_t)b0 * TDIM * HDIM + hcol;

    // ---- x staging: thread -> (row, 4 consecutive F), float4 prefetch ----
    const int xrow = tid >> 4;              // 0..15
    const int xf   = (tid & 15) * 4;        // 0..60
    const float* xp = inp + ((size_t)(b0 + xrow) * TDIM) * FDIM + xf;
    float4 xreg = *(const float4*)(xp);     // x(t=0)

    for (int t = 0; t < TDIM; ++t) {
        const int pr = t & 1;               // read parity
        const int pw = (t + 1) & 1;         // write parity

        // ---- stage x(t) from registers -> LDS (one ds_store_b64) ----
        BF4 xb;
        xb.b[0] = f2bf(xreg.x);
        xb.b[1] = f2bf(xreg.y);
        xb.b[2] = f2bf(xreg.z);
        xb.b[3] = f2bf(xreg.w);
        *(uint2*)&xbuf[pr][xrow * LDX + xf] = xb.u;

        // ---- prefetch x(t+1) into registers (overlaps full compute step) ----
        int tn = (t + 1 < TDIM) ? (t + 1) : t;
        xreg = *(const float4*)(xp + (size_t)tn * FDIM);

        __syncthreads();   // publish x(t) and h(t-1); closes WAR on both buffers

        // ---- A fragments (shared across this wave's 4 gate tiles) ----
        v16bf ax0 = load_afrag_lds(xbuf[pr], LDX, 0,  lane);
        v16bf ax1 = load_afrag_lds(xbuf[pr], LDX, 32, lane);
        v16bf ah[4];
#pragma unroll
        for (int kb = 0; kb < 4; ++kb)
            ah[kb] = load_afrag_lds(hbuf[pr], LDH, kb * 32, lane);

        // ---- z = b + x@Wx + h@Wh : one 16x16 tile per gate per wave ----
        v8f acc[4];
#pragma unroll
        for (int g = 0; g < 4; ++g) {
            v8f a;
#pragma unroll
            for (int r = 0; r < 8; ++r) a[r] = bg[g];
            a = __builtin_amdgcn_wmma_f32_16x16x32_bf16(false, ax0, false, bx[g][0],
                                                        (short)0, a, false, false);
            a = __builtin_amdgcn_wmma_f32_16x16x32_bf16(false, ax1, false, bx[g][1],
                                                        (short)0, a, false, false);
#pragma unroll
            for (int kb = 0; kb < 4; ++kb)
                a = __builtin_amdgcn_wmma_f32_16x16x32_bf16(false, ah[kb], false,
                                                            bh[g][kb], (short)0,
                                                            a, false, false);
            acc[g] = a;
        }

        // ---- pointwise LSTM cell update: acc[0..3] = i,f,g,o (same layout) ----
#pragma unroll
        for (int r = 0; r < 8; ++r) {
            float iv = fast_sigmoid(acc[0][r]);
            float fv = fast_sigmoid(acc[1][r]);
            float gv = fast_tanh(acc[2][r]);
            float ov = fast_sigmoid(acc[3][r]);
            float cn = fv * cst[r] + iv * gv;
            cst[r]   = cn;
            float hv = ov * fast_tanh(cn);
            int mrow = r + 8 * mhi;
            outp[(size_t)mrow * (TDIM * HDIM) + (size_t)t * HDIM] = hv;
            hbuf[pw][mrow * LDH + hcol] = f2bf(hv);   // opposite buffer: no race
        }
        // no second barrier: next iteration's __syncthreads publishes hbuf[pw]
    }
}

extern "C" void kernel_launch(void* const* d_in, const int* in_sizes, int n_in,
                              void* d_out, int out_size, void* d_ws, size_t ws_size,
                              hipStream_t stream) {
    (void)in_sizes; (void)n_in; (void)out_size; (void)d_ws; (void)ws_size;
    const float* inp  = (const float*)d_in[0];  // [B,T,F]
    const float* Wx   = (const float*)d_in[1];  // [F,4H]
    const float* Wh   = (const float*)d_in[2];  // [H,4H]
    const float* bias = (const float*)d_in[3];  // [4H]
    float* out = (float*)d_out;                 // [B,T,H]

    dim3 grid(BDIM / 16);
    dim3 block(256);
    lstm_fused_wmma<<<grid, block, 0, stream>>>(inp, Wx, Wh, bias, out);
}
